// TacotronTTSModel_73718818669302
// MI455X (gfx1250) — matrix operand order repaired and checked by reading, not compiled, not run on hardware
//
#include <hip/hip_runtime.h>
#include <math.h>

typedef unsigned short u16t;
typedef __attribute__((ext_vector_type(8)))  u16t   u16x8;
typedef __attribute__((ext_vector_type(16))) u16t   u16x16;
typedef __attribute__((ext_vector_type(16))) __bf16 bf16x16;
typedef __attribute__((ext_vector_type(8)))  float  f32x8;

#define DEVINL __device__ __forceinline__

DEVINL u16t f2bf(float f) {
  unsigned u = __float_as_uint(f);
  if ((u & 0x7F800000u) == 0x7F800000u) return (u16t)(u >> 16);  // Inf/NaN passthrough
  unsigned r = u + 0x7FFFu + ((u >> 16) & 1u);                    // RNE
  return (u16t)(r >> 16);
}

DEVINL float sigm(float x) { return 1.0f / (1.0f + __expf(-x)); }

DEVINL f32x8 wmma_bf16(bf16x16 a, bf16x16 b, f32x8 c) {
  return __builtin_amdgcn_wmma_f32_16x16x32_bf16(false, a, false, b, (short)0, c, false, false);
}

// A fragment: 16x32 bf16 tile (M x K), A row-major with leading dim lda (elements).
// Lane 0-15: row=lane,  K chunks {k0+0..7,  k0+16..23}
// Lane16-31: row=lane-16,K chunks {k0+8..15, k0+24..31}
DEVINL bf16x16 frag_a(const u16t* A, long long lda, int row0, int k0, int lane, int maxRow) {
  int r = row0 + (lane & 15);
  r = r < maxRow ? r : (maxRow - 1);
  int kb = k0 + ((lane & 16) ? 8 : 0);
  const u16t* p = A + (long long)r * lda + kb;
  u16x8 lo = *(const u16x8*)(p);
  u16x8 hi = *(const u16x8*)(p + 16);
  u16x16 v;
#pragma unroll
  for (int i = 0; i < 8; ++i) { v[i] = lo[i]; v[i + 8] = hi[i]; }
  return __builtin_bit_cast(bf16x16, v);
}

// Same as frag_a but row0 may be shifted out of [0,L): those rows contribute zero (conv padding).
DEVINL bf16x16 frag_a_shift(const u16t* A, long long lda, int row0, int k0, int lane, int L) {
  int r = row0 + (lane & 15);
  bool ok = (r >= 0) && (r < L);
  int rc = ok ? r : 0;
  int kb = k0 + ((lane & 16) ? 8 : 0);
  const u16t* p = A + (long long)rc * lda + kb;
  u16x8 lo = *(const u16x8*)(p);
  u16x8 hi = *(const u16x8*)(p + 16);
  u16x16 v;
#pragma unroll
  for (int i = 0; i < 8; ++i) {
    v[i]     = ok ? lo[i] : (u16t)0;
    v[i + 8] = ok ? hi[i] : (u16t)0;
  }
  return __builtin_bit_cast(bf16x16, v);
}

// B fragment: 32x16 bf16 tile (K x N) sourced from W[N,K] row-major (so B = W^T).
// Lane 0-15: col=lane,   K chunks {k0+0..7, k0+16..23} ; Lane16-31: col=lane-16, K chunks {k0+8..15, k0+24..31}
DEVINL bf16x16 frag_b(const u16t* W, long long ldw, int col0, int k0, int lane, int maxCol) {
  int n = col0 + (lane & 15);
  n = n < maxCol ? n : (maxCol - 1);
  int kb = k0 + ((lane & 16) ? 8 : 0);
  const u16t* p = W + (long long)n * ldw + kb;
  u16x8 lo = *(const u16x8*)(p);
  u16x8 hi = *(const u16x8*)(p + 16);
  u16x16 v;
#pragma unroll
  for (int i = 0; i < 8; ++i) { v[i] = lo[i]; v[i + 8] = hi[i]; }
  return __builtin_bit_cast(bf16x16, v);
}

// -------------------- generic batched GEMM: C = A @ W^T (+bias) --------------------
// batch index z decomposed as (zb, zh) with zh in [0,nh)
__global__ __launch_bounds__(256) void k_gemm(
    const u16t* __restrict__ A, const u16t* __restrict__ W,
    const float* __restrict__ bias, float* __restrict__ C,
    int M, int N, int K, int lda, int ldb, int ldc, int nh,
    long long sAb, long long sAh, long long sBb, long long sBh,
    long long sCb, long long sCh) {
  int z = blockIdx.z;
  int zb = z / nh, zh = z - zb * nh;
  A += zb * sAb + zh * sAh;
  W += zb * sBb + zh * sBh;
  C += zb * sCb + zh * sCh;
  int lane = threadIdx.x & 31;
  int wave = blockIdx.x * (blockDim.x >> 5) + (threadIdx.x >> 5);
  int ntiles = (N + 15) >> 4;
  int mtiles = (M + 15) >> 4;
  if (wave >= ntiles * mtiles) return;
  int tm = (wave / ntiles) << 4;
  int tn = (wave % ntiles) << 4;
  int col = tn + (lane & 15);
  float bv = bias ? bias[col < N ? col : (N - 1)] : 0.0f;
  f32x8 acc;
#pragma unroll
  for (int j = 0; j < 8; ++j) acc[j] = bv;
  for (int k0 = 0; k0 < K; k0 += 32) {
    if (k0 + 32 < K)
      __builtin_prefetch(W + (long long)(tn + (lane & 15)) * ldb + k0 + 32, 0, 3);
    bf16x16 a = frag_a(A, lda, tm, k0, lane, M);
    bf16x16 b = frag_b(W, ldb, tn, k0, lane, N);
    acc = wmma_bf16(a, b, acc);
  }
  int rb = tm + ((lane & 16) ? 8 : 0);
#pragma unroll
  for (int j = 0; j < 8; ++j) {
    int r = rb + j;
    if (r < M && col < N) C[(long long)r * ldc + col] = acc[j];
  }
}

// -------------------- postnet conv as 5 shifted GEMMs --------------------
// X: [L, Cp] bf16 per batch (zero-padded channels), Wt: [5][O][Cp] bf16, Y: [L, O] f32
__global__ __launch_bounds__(256) void k_conv(
    const u16t* __restrict__ X, const u16t* __restrict__ Wt,
    const float* __restrict__ bias, const float* __restrict__ resid,
    float* __restrict__ Y, int L, int O, int Cp,
    long long sX, long long sY, long long sR) {
  int b = blockIdx.z;
  X += (long long)b * sX;
  Y += (long long)b * sY;
  const float* rz = resid ? resid + (long long)b * sR : nullptr;
  int lane = threadIdx.x & 31;
  int wave = blockIdx.x * (blockDim.x >> 5) + (threadIdx.x >> 5);
  int ntiles = O >> 4;
  int mtiles = (L + 15) >> 4;
  if (wave >= ntiles * mtiles) return;
  int tm = (wave / ntiles) << 4;
  int tn = (wave % ntiles) << 4;
  int col = tn + (lane & 15);
  float bv = bias[col];
  f32x8 acc;
#pragma unroll
  for (int j = 0; j < 8; ++j) acc[j] = bv;
#pragma unroll
  for (int dt = 0; dt < 5; ++dt) {
    const u16t* Wd = Wt + (long long)dt * O * Cp;
    int row0 = tm + dt - 2;
    for (int k0 = 0; k0 < Cp; k0 += 32) {
      bf16x16 a = frag_a_shift(X, Cp, row0, k0, lane, L);
      bf16x16 w = frag_b(Wd, Cp, tn, k0, lane, O);
      acc = wmma_bf16(a, w, acc);
    }
  }
  int rb = tm + ((lane & 16) ? 8 : 0);
#pragma unroll
  for (int j = 0; j < 8; ++j) {
    int r = rb + j;
    if (r < L) {
      float v = acc[j];
      if (rz) v += rz[(long long)r * O + col];
      Y[(long long)r * O + col] = v;
    }
  }
}

// -------------------- bidirectional encoder LSTM (persistent, 1 WG per direction) ----
// blockIdx.x: 0=forward, 1=backward. 512 threads = 16 waves, 2 tasks/wave.
// LDS: c[32*256] f32 + hbf[32*256] bf16.  G: [B,S,1024] precomputed input gates+bias.
__global__ __launch_bounds__(512) void k_enc_lstm(
    const u16t* __restrict__ whh_f, const u16t* __restrict__ whh_b,
    const float* __restrict__ Gf, const float* __restrict__ Gb,
    u16t* __restrict__ encbf) {
  extern __shared__ char smem[];
  float* c_s = (float*)smem;                       // 32*256 f32
  u16t* hbf  = (u16t*)(smem + 32 * 256 * 4);       // 32*256 bf16
  int dir = blockIdx.x;
  const u16t* whh = dir ? whh_b : whh_f;
  const float* G  = dir ? Gb : Gf;
  int tid = threadIdx.x, lane = tid & 31, wave = tid >> 5;
  for (int i = tid; i < 32 * 256; i += blockDim.x) { c_s[i] = 0.f; hbf[i] = 0; }
  __syncthreads();
  for (int step = 0; step < 256; ++step) {
    int s = dir ? (255 - step) : step;
    f32x8 accs[2][4];
    int tms[2], tns[2];
#pragma unroll
    for (int q = 0; q < 2; ++q) {
      int task = wave * 2 + q;          // 0..31
      int tm = (task >> 4) << 4;        // batch tile 0/16
      int tn = (task & 15) << 4;        // hidden-unit tile
      tms[q] = tm; tns[q] = tn;
      int colu = tn + (lane & 15);
      int rb = tm + ((lane & 16) ? 8 : 0);
#pragma unroll
      for (int g = 0; g < 4; ++g) {
        f32x8 a;
#pragma unroll
        for (int j = 0; j < 8; ++j)
          a[j] = G[((long long)(rb + j) * 256 + s) * 1024 + g * 256 + colu];
        accs[q][g] = a;
      }
      for (int k0 = 0; k0 < 256; k0 += 32) {
        bf16x16 af = frag_a(hbf, 256, tm, k0, lane, 32);
#pragma unroll
        for (int g = 0; g < 4; ++g) {
          bf16x16 bw = frag_b(whh, 256, g * 256 + tn, k0, lane, 1024);
          accs[q][g] = wmma_bf16(af, bw, accs[q][g]);
        }
      }
    }
    __syncthreads();                   // all hbf reads done
#pragma unroll
    for (int q = 0; q < 2; ++q) {
      int tm = tms[q], tn = tns[q];
      int colu = tn + (lane & 15);
      int rb = tm + ((lane & 16) ? 8 : 0);
#pragma unroll
      for (int j = 0; j < 8; ++j) {
        int m = rb + j;
        float cn = sigm(accs[q][1][j]) * c_s[m * 256 + colu]
                 + sigm(accs[q][0][j]) * tanhf(accs[q][2][j]);
        float hn = sigm(accs[q][3][j]) * tanhf(cn);
        c_s[m * 256 + colu] = cn;
        u16t hb = f2bf(hn);
        hbf[m * 256 + colu] = hb;
        encbf[((long long)m * 256 + s) * 512 + dir * 256 + colu] = hb;
      }
    }
    __syncthreads();
  }
}

// -------------------- decoder LSTM (persistent, single WG, 32 waves) --------------------
// Per step: gates = cgate(const, incl. biases+ctx part) + mel_in@Wmel^T + h@Whh^T
// LDS: c[32*512] f32 + hbf[32*512] bf16.  melin: [B,T,96] bf16. dh out: [B,T,512] bf16.
__global__ __launch_bounds__(1024) void k_dec_lstm(
    const u16t* __restrict__ whh, const u16t* __restrict__ wmel,
    const u16t* __restrict__ melin, const float* __restrict__ cgate,
    u16t* __restrict__ dh) {
  extern __shared__ char smem[];
  float* c_s = (float*)smem;                       // 32*512 f32
  u16t* hbf  = (u16t*)(smem + 32 * 512 * 4);       // 32*512 bf16
  int tid = threadIdx.x, lane = tid & 31, wave = tid >> 5;  // 32 waves
  for (int i = tid; i < 32 * 512; i += blockDim.x) { c_s[i] = 0.f; hbf[i] = 0; }
  __syncthreads();
  for (int t = 0; t < 1000; ++t) {
    const u16t* Amel = melin + (long long)t * 96;  // rows stride T*96
    f32x8 accs[2][4];
    int tms[2], tns[2];
#pragma unroll
    for (int q = 0; q < 2; ++q) {
      int task = wave * 2 + q;          // 0..63
      int tm = (task >> 5) << 4;
      int tn = (task & 31) << 4;
      tms[q] = tm; tns[q] = tn;
      int colu = tn + (lane & 15);
      int rb = tm + ((lane & 16) ? 8 : 0);
#pragma unroll
      for (int g = 0; g < 4; ++g) {
        f32x8 a;
#pragma unroll
        for (int j = 0; j < 8; ++j)
          a[j] = cgate[(rb + j) * 2048 + g * 512 + colu];
        accs[q][g] = a;
      }
      for (int k0 = 0; k0 < 96; k0 += 32) {        // mel input part (K padded 80->96)
        bf16x16 af = frag_a(Amel, 96000, tm, k0, lane, 32);
#pragma unroll
        for (int g = 0; g < 4; ++g) {
          bf16x16 bw = frag_b(wmel, 96, g * 512 + tn, k0, lane, 2048);
          accs[q][g] = wmma_bf16(af, bw, accs[q][g]);
        }
      }
      for (int k0 = 0; k0 < 512; k0 += 32) {       // recurrent part
        bf16x16 af = frag_a(hbf, 512, tm, k0, lane, 32);
#pragma unroll
        for (int g = 0; g < 4; ++g) {
          bf16x16 bw = frag_b(whh, 512, g * 512 + tn, k0, lane, 2048);
          accs[q][g] = wmma_bf16(af, bw, accs[q][g]);
        }
      }
    }
    __syncthreads();
#pragma unroll
    for (int q = 0; q < 2; ++q) {
      int tm = tms[q], tn = tns[q];
      int colu = tn + (lane & 15);
      int rb = tm + ((lane & 16) ? 8 : 0);
#pragma unroll
      for (int j = 0; j < 8; ++j) {
        int m = rb + j;
        float cn = sigm(accs[q][1][j]) * c_s[m * 512 + colu]
                 + sigm(accs[q][0][j]) * tanhf(accs[q][2][j]);
        float hn = sigm(accs[q][3][j]) * tanhf(cn);
        c_s[m * 512 + colu] = cn;
        u16t hb = f2bf(hn);
        hbf[m * 512 + colu] = hb;
        dh[((long long)m * 1000 + t) * 512 + colu] = hb;
      }
    }
    __syncthreads();
  }
}

// -------------------- elementwise / small kernels --------------------
__global__ void k_cvt(const float* __restrict__ src, u16t* __restrict__ dst,
                      long long rows, int cols, int srcLd, int srcOff, int dstLd) {
  long long i = (long long)blockIdx.x * blockDim.x + threadIdx.x;
  long long total = rows * (long long)dstLd;
  if (i >= total) return;
  long long r = i / dstLd;
  int c = (int)(i - r * dstLd);
  dst[i] = (c < cols) ? f2bf(src[r * srcLd + srcOff + c]) : (u16t)0;
}

__global__ void k_cvt_convw(const float* __restrict__ w, u16t* __restrict__ wt,
                            int O, int C, int Cp) {
  int i = blockIdx.x * blockDim.x + threadIdx.x;
  int total = 5 * O * Cp;
  if (i >= total) return;
  int c = i % Cp;
  int rest = i / Cp;
  int o = rest % O;
  int dt = rest / O;
  wt[i] = (c < C) ? f2bf(w[(o * C + c) * 5 + dt]) : (u16t)0;
}

__global__ void k_embed(const int* __restrict__ tok, const float* __restrict__ emb,
                        u16t* __restrict__ xbf) {
  int i = blockIdx.x * blockDim.x + threadIdx.x;
  if (i >= 32 * 256 * 512) return;
  int bs = i >> 9, k = i & 511;
  xbf[i] = f2bf(emb[(long long)tok[bs] * 512 + k]);
}

__global__ void k_melin(const float* __restrict__ mel, u16t* __restrict__ out) {
  int i = blockIdx.x * blockDim.x + threadIdx.x;
  if (i >= 32 * 1000 * 96) return;
  int c = i % 96;
  int bt = i / 96;
  int t = bt % 1000, b = bt / 1000;
  float v = (t == 0 || c >= 80) ? 0.f : mel[((long long)b * 1000 + (t - 1)) * 80 + c];
  out[i] = f2bf(v);
}

__global__ void k_biassum(const float* __restrict__ a, const float* __restrict__ b,
                          float* __restrict__ o, int n) {
  int i = blockIdx.x * blockDim.x + threadIdx.x;
  if (i < n) o[i] = a[i] + b[i];
}

__global__ void k_tanhcvt(const float* __restrict__ s, u16t* __restrict__ d, long long n) {
  long long i = (long long)blockIdx.x * blockDim.x + threadIdx.x;
  if (i < n) d[i] = f2bf(tanhf(s[i]));
}

// softmax in place over last dim (256) with 1/sqrt(d)=0.125 pre-scale; one wave per row
__global__ void k_softmax(float* __restrict__ S_, int rows) {
  int wave = blockIdx.x * (blockDim.x >> 5) + (threadIdx.x >> 5);
  int lane = threadIdx.x & 31;
  if (wave >= rows) return;
  float* row = S_ + (long long)wave * 256;
  float v[8];
  float mx = -1e30f;
#pragma unroll
  for (int i = 0; i < 8; ++i) { v[i] = row[lane + 32 * i] * 0.125f; mx = fmaxf(mx, v[i]); }
  for (int m = 16; m > 0; m >>= 1) mx = fmaxf(mx, __shfl_xor(mx, m));
  float sum = 0.f;
#pragma unroll
  for (int i = 0; i < 8; ++i) { v[i] = __expf(v[i] - mx); sum += v[i]; }
  for (int m = 16; m > 0; m >>= 1) sum += __shfl_xor(sum, m);
  float inv = 1.0f / sum;
#pragma unroll
  for (int i = 0; i < 8; ++i) row[lane + 32 * i] = v[i] * inv;
}

// pbar[bh,k] = mean_s probs[bh,s,k]
__global__ void k_colmean(const float* __restrict__ P, float* __restrict__ pbar) {
  int bh = blockIdx.x, k = threadIdx.x;
  const float* base = P + (long long)bh * 65536;
  float s = 0.f;
  for (int si = 0; si < 256; ++si) s += base[si * 256 + k];
  pbar[bh * 256 + k] = s * (1.0f / 256.0f);
}

// ctx_pre[b, h*64+d] = sum_k pbar[b,h,k] * V[b,k,h*64+d]  (V = qkv[..., 1024:1536])
__global__ void k_ctxpre(const float* __restrict__ pbar, const float* __restrict__ qkv,
                         float* __restrict__ out) {
  int idx = blockIdx.x * blockDim.x + threadIdx.x;
  if (idx >= 32 * 512) return;
  int b = idx >> 9, hd = idx & 511;
  int h = hd >> 6;
  const float* pb = pbar + (b * 8 + h) * 256;
  float s = 0.f;
  for (int k = 0; k < 256; ++k)
    s += pb[k] * qkv[(long long)(b * 256 + k) * 1536 + 1024 + hd];
  out[idx] = s;
}

// =====================================================================================
extern "C" void kernel_launch(void* const* d_in, const int* in_sizes, int n_in,
                              void* d_out, int out_size, void* d_ws, size_t ws_size,
                              hipStream_t stream) {
  (void)in_sizes; (void)n_in; (void)out_size; (void)ws_size;
  auto cdiv = [](long long a, long long b) { return (int)((a + b - 1) / b); };

  // ---- inputs (setup_inputs order) ----
  const int*   tok      = (const int*)d_in[0];
  const float* mel_tgt  = (const float*)d_in[1];
  const float* emb      = (const float*)d_in[3];
  const float* ef_wih   = (const float*)d_in[4];
  const float* ef_whh   = (const float*)d_in[5];
  const float* ef_bih   = (const float*)d_in[6];
  const float* ef_bhh   = (const float*)d_in[7];
  const float* eb_wih   = (const float*)d_in[8];
  const float* eb_whh   = (const float*)d_in[9];
  const float* eb_bih   = (const float*)d_in[10];
  const float* eb_bhh   = (const float*)d_in[11];
  const float* wqkv     = (const float*)d_in[12];
  const float* bqkv     = (const float*)d_in[13];
  const float* wo       = (const float*)d_in[14];
  const float* bo       = (const float*)d_in[15];
  const float* d_wih    = (const float*)d_in[16];
  const float* d_whh    = (const float*)d_in[17];
  const float* d_bih    = (const float*)d_in[18];
  const float* d_bhh    = (const float*)d_in[19];
  const float* mel_w    = (const float*)d_in[20];
  const float* mel_b    = (const float*)d_in[21];
  const float* stop_w   = (const float*)d_in[22];
  const float* stop_b   = (const float*)d_in[23];
  const float* pn1_w    = (const float*)d_in[24];
  const float* pn1_b    = (const float*)d_in[25];
  const float* pn2_w    = (const float*)d_in[26];
  const float* pn2_b    = (const float*)d_in[27];
  const float* pn3_w    = (const float*)d_in[28];
  const float* pn3_b    = (const float*)d_in[29];

  float* mel_out  = (float*)d_out;             // [32,1000,80]
  float* melpost  = mel_out + 2560000;         // [32,1000,80]
  float* stop_out = mel_out + 5120000;         // [32,1000,1]

  // ---- workspace carve-up (256B aligned) ----
  size_t off = 0;
  auto A = [&](size_t bytes) {
    off = (off + 255) & ~(size_t)255;
    char* p = (char*)d_ws + off;
    off += bytes;
    return (void*)p;
  };
  u16t* xbf      = (u16t*)A((size_t)32 * 256 * 512 * 2);
  u16t* wihf_bf  = (u16t*)A(1024 * 512 * 2);
  u16t* whhf_bf  = (u16t*)A(1024 * 256 * 2);
  u16t* wihb_bf  = (u16t*)A(1024 * 512 * 2);
  u16t* whhb_bf  = (u16t*)A(1024 * 256 * 2);
  u16t* wqkv_bf  = (u16t*)A(1536 * 512 * 2);
  u16t* wo_bf    = (u16t*)A(512 * 512 * 2);
  u16t* dwhh_bf  = (u16t*)A(2048 * 512 * 2);
  u16t* wmel_bf  = (u16t*)A(2048 * 96 * 2);
  u16t* wctx_bf  = (u16t*)A(2048 * 512 * 2);
  u16t* melw_bf  = (u16t*)A(80 * 512 * 2);
  u16t* stopw_bf = (u16t*)A(512 * 2);
  u16t* pn1wt    = (u16t*)A(5 * 512 * 96 * 2);
  u16t* pn2wt    = (u16t*)A((size_t)5 * 512 * 512 * 2);
  u16t* pn3wt    = (u16t*)A(5 * 80 * 512 * 2);
  float* bsumf   = (float*)A(1024 * 4);
  float* bsumb   = (float*)A(1024 * 4);
  float* bsumd   = (float*)A(2048 * 4);
  float* Gf      = (float*)A((size_t)32 * 256 * 1024 * 4);
  float* Gb      = (float*)A((size_t)32 * 256 * 1024 * 4);
  u16t* encbf    = (u16t*)A((size_t)32 * 256 * 512 * 2);
  float* qkv     = (float*)A((size_t)32 * 256 * 1536 * 4);
  u16t* qkvbf    = (u16t*)A((size_t)32 * 256 * 1536 * 2);
  float* scores  = (float*)A((size_t)32 * 8 * 256 * 256 * 4);
  float* pbar    = (float*)A(32 * 8 * 256 * 4);
  float* ctxpre  = (float*)A(32 * 512 * 4);
  u16t* ctxprebf = (u16t*)A(32 * 512 * 2);
  float* ctx     = (float*)A(32 * 512 * 4);
  u16t* ctxbf    = (u16t*)A(32 * 512 * 2);
  float* cgate   = (float*)A(32 * 2048 * 4);
  u16t* melinbf  = (u16t*)A((size_t)32 * 1000 * 96 * 2);
  u16t* dhbf     = (u16t*)A((size_t)32 * 1000 * 512 * 2);
  u16t* mel96    = (u16t*)A((size_t)32 * 1000 * 96 * 2);
  float* ybuf    = (float*)A((size_t)32 * 1000 * 512 * 4);
  u16t* x2bf     = (u16t*)A((size_t)32 * 1000 * 512 * 2);
  u16t* x3bf     = (u16t*)A((size_t)32 * 1000 * 512 * 2);

  const int TPB = 256;
  auto gemm = [&](const u16t* Ap, const u16t* Wp, const float* bp, float* Cp,
                  int M, int N, int K, int lda, int ldb, int ldc,
                  int batch, int nh, long long sAb, long long sAh,
                  long long sBb, long long sBh, long long sCb, long long sCh) {
    int tiles = cdiv(M, 16) * cdiv(N, 16);
    dim3 g(cdiv(tiles, 8), 1, batch);
    hipLaunchKernelGGL(k_gemm, g, dim3(TPB), 0, stream, Ap, Wp, bp, Cp,
                       M, N, K, lda, ldb, ldc, nh, sAb, sAh, sBb, sBh, sCb, sCh);
  };

  // ---- phase 0: conversions / precompute ----
  k_embed<<<cdiv((long long)32 * 256 * 512, TPB), TPB, 0, stream>>>(tok, emb, xbf);
  k_cvt<<<cdiv(1024 * 512, TPB), TPB, 0, stream>>>(ef_wih, wihf_bf, 1024, 512, 512, 0, 512);
  k_cvt<<<cdiv(1024 * 256, TPB), TPB, 0, stream>>>(ef_whh, whhf_bf, 1024, 256, 256, 0, 256);
  k_cvt<<<cdiv(1024 * 512, TPB), TPB, 0, stream>>>(eb_wih, wihb_bf, 1024, 512, 512, 0, 512);
  k_cvt<<<cdiv(1024 * 256, TPB), TPB, 0, stream>>>(eb_whh, whhb_bf, 1024, 256, 256, 0, 256);
  k_cvt<<<cdiv(1536 * 512, TPB), TPB, 0, stream>>>(wqkv, wqkv_bf, 1536, 512, 512, 0, 512);
  k_cvt<<<cdiv(512 * 512, TPB), TPB, 0, stream>>>(wo, wo_bf, 512, 512, 512, 0, 512);
  k_cvt<<<cdiv(2048 * 512, TPB), TPB, 0, stream>>>(d_whh, dwhh_bf, 2048, 512, 512, 0, 512);
  k_cvt<<<cdiv(2048 * 96, TPB), TPB, 0, stream>>>(d_wih, wmel_bf, 2048, 80, 592, 0, 96);
  k_cvt<<<cdiv(2048 * 512, TPB), TPB, 0, stream>>>(d_wih, wctx_bf, 2048, 512, 592, 80, 512);
  k_cvt<<<cdiv(80 * 512, TPB), TPB, 0, stream>>>(mel_w, melw_bf, 80, 512, 512, 0, 512);
  k_cvt<<<cdiv(512, TPB), TPB, 0, stream>>>(stop_w, stopw_bf, 1, 512, 512, 0, 512);
  k_cvt_convw<<<cdiv(5 * 512 * 96, TPB), TPB, 0, stream>>>(pn1_w, pn1wt, 512, 80, 96);
  k_cvt_convw<<<cdiv(5 * 512 * 512, TPB), TPB, 0, stream>>>(pn2_w, pn2wt, 512, 512, 512);
  k_cvt_convw<<<cdiv(5 * 80 * 512, TPB), TPB, 0, stream>>>(pn3_w, pn3wt, 80, 512, 512);
  k_biassum<<<cdiv(1024, TPB), TPB, 0, stream>>>(ef_bih, ef_bhh, bsumf, 1024);
  k_biassum<<<cdiv(1024, TPB), TPB, 0, stream>>>(eb_bih, eb_bhh, bsumb, 1024);
  k_biassum<<<cdiv(2048, TPB), TPB, 0, stream>>>(d_bih, d_bhh, bsumd, 2048);
  k_melin<<<cdiv((long long)32 * 1000 * 96, TPB), TPB, 0, stream>>>(mel_tgt, melinbf);

  // ---- encoder: input gates (batched GEMM) then recurrence ----
  gemm(xbf, wihf_bf, bsumf, Gf, 8192, 1024, 512, 512, 512, 1024, 1, 1, 0, 0, 0, 0, 0, 0);
  gemm(xbf, wihb_bf, bsumb, Gb, 8192, 1024, 512, 512, 512, 1024, 1, 1, 0, 0, 0, 0, 0, 0);
  k_enc_lstm<<<2, 512, 32 * 256 * 4 + 32 * 256 * 2, stream>>>(whhf_bf, whhb_bf, Gf, Gb, encbf);

  // ---- attention (hoisted): qkv, scores, softmax, column-mean, ctx ----
  gemm(encbf, wqkv_bf, bqkv, qkv, 8192, 1536, 512, 512, 512, 1536, 1, 1, 0, 0, 0, 0, 0, 0);
  k_cvt<<<cdiv((long long)8192 * 1536, TPB), TPB, 0, stream>>>(qkv, qkvbf, 8192, 1536, 1536, 0, 1536);
  gemm(qkvbf, qkvbf + 512, nullptr, scores, 256, 256, 64, 1536, 1536, 256,
       256, 8, 256LL * 1536, 64, 256LL * 1536, 64, 8LL * 65536, 65536);
  k_softmax<<<cdiv(65536, 8), TPB, 0, stream>>>(scores, 65536);
  k_colmean<<<256, 256, 0, stream>>>(scores, pbar);
  k_ctxpre<<<cdiv(32 * 512, TPB), TPB, 0, stream>>>(pbar, qkv, ctxpre);
  k_cvt<<<cdiv(32 * 512, TPB), TPB, 0, stream>>>(ctxpre, ctxprebf, 32, 512, 512, 0, 512);
  gemm(ctxprebf, wo_bf, bo, ctx, 32, 512, 512, 512, 512, 512, 1, 1, 0, 0, 0, 0, 0, 0);
  k_cvt<<<cdiv(32 * 512, TPB), TPB, 0, stream>>>(ctx, ctxbf, 32, 512, 512, 0, 512);
  gemm(ctxbf, wctx_bf, bsumd, cgate, 32, 2048, 512, 512, 512, 2048, 1, 1, 0, 0, 0, 0, 0, 0);

  // ---- decoder recurrence ----
  k_dec_lstm<<<1, 1024, 32 * 512 * 4 + 32 * 512 * 2, stream>>>(dwhh_bf, wmel_bf, melinbf,
                                                               cgate, dhbf);

  // ---- projections ----
  gemm(dhbf, melw_bf, mel_b, mel_out, 32000, 80, 512, 512, 512, 80, 1, 1, 0, 0, 0, 0, 0, 0);
  gemm(dhbf, stopw_bf, stop_b, stop_out, 32000, 1, 512, 512, 512, 1, 1, 1, 0, 0, 0, 0, 0, 0);

  // ---- postnet ----
  k_cvt<<<cdiv((long long)32000 * 96, TPB), TPB, 0, stream>>>(mel_out, mel96, 32000, 80, 80, 0, 96);
  {
    int tiles = cdiv(1000, 16) * (512 / 16);
    dim3 g(cdiv(tiles, 8), 1, 32);
    k_conv<<<g, TPB, 0, stream>>>(mel96, pn1wt, pn1_b, nullptr, ybuf, 1000, 512, 96,
                                  96000LL, 512000LL, 0LL);
  }
  k_tanhcvt<<<cdiv((long long)32 * 1000 * 512, TPB), TPB, 0, stream>>>(ybuf, x2bf,
                                                                       (long long)32 * 1000 * 512);
  {
    int tiles = cdiv(1000, 16) * (512 / 16);
    dim3 g(cdiv(tiles, 8), 1, 32);
    k_conv<<<g, TPB, 0, stream>>>(x2bf, pn2wt, pn2_b, nullptr, ybuf, 1000, 512, 512,
                                  512000LL, 512000LL, 0LL);
  }
  k_tanhcvt<<<cdiv((long long)32 * 1000 * 512, TPB), TPB, 0, stream>>>(ybuf, x3bf,
                                                                       (long long)32 * 1000 * 512);
  {
    int tiles = cdiv(1000, 16) * (80 / 16);
    dim3 g(cdiv(tiles, 8), 1, 32);
    k_conv<<<g, TPB, 0, stream>>>(x3bf, pn3wt, pn3_b, mel_out, melpost, 1000, 80, 512,
                                  512000LL, 80000LL, 80000LL);
  }
}
